// FluxSingleAttention_20401094656236
// MI455X (gfx1250) — matrix operand order repaired and checked by reading, not compiled, not run on hardware
//
#include <hip/hip_runtime.h>
#include <hip/hip_bf16.h>
#include <math.h>

// ---------------------------------------------------------------------------
// FluxSingleAttention for MI455X (gfx1250, wave32, WMMA bf16, async LDS copy)
// B=1, S=2048, D=3072, H=24, HD=128
// ---------------------------------------------------------------------------

#define S_LEN 2048
#define D_DIM 3072
#define H_NUM 24
#define HD_DIM 128

typedef __attribute__((ext_vector_type(16))) __bf16 v16bf;
typedef __attribute__((ext_vector_type(8)))  float  v8f;
typedef __attribute__((ext_vector_type(4)))  __bf16 v4bf;
typedef __attribute__((ext_vector_type(4)))  int    v4i;

union U16 { v16bf v; uint4 q[2]; };

#if defined(__has_builtin)
#if __has_builtin(__builtin_amdgcn_global_load_async_to_lds_b128)
#define USE_ASYNC_COPY 1
#endif
#if __has_builtin(__builtin_amdgcn_s_wait_asynccnt)
#define HAVE_WAIT_ASYNC 1
#endif
#endif

// Copy 16 bytes global -> LDS. Async (ASYNCcnt-tracked, no VGPR round trip)
// when the gfx1250 builtin is available; sync load+ds_store otherwise.
// Builtin signature (from hipcc diagnostic): (v4i AS1*, v4i AS3*, Ii, Ii).
static __device__ inline void cp16(const __bf16* g, __bf16* l) {
#ifdef USE_ASYNC_COPY
    __builtin_amdgcn_global_load_async_to_lds_b128(
        (__attribute__((address_space(1))) v4i*)g,
        (__attribute__((address_space(3))) v4i*)l, 0, 0);
#else
    *(uint4*)l = *(const uint4*)g;
#endif
}

// Wait for this wave's staged copies, then workgroup barrier.
static __device__ inline void stage_wait() {
#ifdef USE_ASYNC_COPY
#ifdef HAVE_WAIT_ASYNC
    __builtin_amdgcn_s_wait_asynccnt(0);
#else
    asm volatile("s_wait_asynccnt 0x0" ::: "memory");
#endif
#endif
    __syncthreads();
}

static __device__ inline v8f wmma_bf16(v16bf a, v16bf b, v8f c) {
    // D = A(16x32 bf16) x B(32x16 bf16) + C(16x16 f32)
    return __builtin_amdgcn_wmma_f32_16x16x32_bf16(
        /*neg_a=*/false, a, /*neg_b=*/false, b,
        /*c_mod=*/(short)0, c, /*reuse_a=*/false, /*reuse_b=*/false);
}

// ---------------------------------------------------------------------------
// Kernel 1: f32 -> bf16 convert (vectorized x4)
// ---------------------------------------------------------------------------
__global__ __launch_bounds__(256) void cvt4_kernel(const float* __restrict__ in,
                                                   __bf16* __restrict__ out, int n4) {
    int i = blockIdx.x * 256 + threadIdx.x;
    if (i < n4) {
        float4 f = ((const float4*)in)[i];
        v4bf o;
        o[0] = (__bf16)f.x; o[1] = (__bf16)f.y;
        o[2] = (__bf16)f.z; o[3] = (__bf16)f.w;
        ((v4bf*)out)[i] = o;
    }
}

// ---------------------------------------------------------------------------
// Kernel 2: QKV projection GEMM (bf16 WMMA, f32 accum) + bias
//           + RMSNorm + RoPE (q,k) or transpose-pack (v)
// grid (32, 24, 3): x = 64-row block of S, y = head, z = {q,k,v}
// block 128 = 4 waves; wave w owns rows [Mbase, Mbase+16), Mbase = 64*blockIdx.x + 16w.
// Weight tile (shared across all 4 waves) is double-buffered in LDS via
// async global->LDS copies. All 8 B-fragments are loaded up-front into
// distinct register groups so the DS waits before the WMMAs are partial
// (in-order DS returns), overlapping WMMA t with loads for t+1..7.
// ---------------------------------------------------------------------------
__global__ __launch_bounds__(128) void qkv_kernel(
    const __bf16* __restrict__ xb,
    const __bf16* __restrict__ Wqb, const __bf16* __restrict__ Wkb, const __bf16* __restrict__ Wvb,
    const float* __restrict__ bq, const float* __restrict__ bk, const float* __restrict__ bv,
    const float* __restrict__ gq, const float* __restrict__ gk,
    const float* __restrict__ cosb, const float* __restrict__ sinb,
    __bf16* __restrict__ qo, __bf16* __restrict__ ko, __bf16* __restrict__ vTo) {

    const int tid = threadIdx.x;
    const int l   = tid & 31;
    const int w   = tid >> 5;
    const int lo  = l & 15;
    const int hi  = l >> 4;
    const int head = blockIdx.y;
    const int z    = blockIdx.z;
    const int Mbase = blockIdx.x * 64 + w * 16;

    const __bf16* W    = (z == 0) ? Wqb : ((z == 1) ? Wkb : Wvb);
    const float*  bias = (z == 0) ? bq  : ((z == 1) ? bk  : bv);

    // B tile stage: [col 0..127][k 0..31] bf16, double-buffered (2 x 8KB)
    __shared__ __bf16 bstage[2][128 * 32];
    __shared__ float  smem[64 * 128];
    __shared__ float  sinv[64];

    const __bf16* Whead = W + (size_t)head * HD_DIM * D_DIM;

    // Stage one 128x32 weight tile: 512 16B chunks, 4 per thread.
    auto load_btile = [&](int kb, int p) {
#pragma unroll
        for (int j = 0; j < 4; ++j) {
            const int chunk = tid + 128 * j;   // 0..511
            const int col   = chunk >> 2;      // output column within head
            const int part  = chunk & 3;       // 8-half sub-chunk
            cp16(Whead + (size_t)col * D_DIM + kb + part * 8,
                 &bstage[p][col * 32 + part * 8]);
        }
    };

    v8f acc[8];
#pragma unroll
    for (int t = 0; t < 8; ++t) acc[t] = (v8f){};

    // A: row (Mbase+lo) of x, K-contiguous.
    const __bf16* arow = xb + (size_t)(Mbase + lo) * D_DIM;

    load_btile(0, 0);
    for (int kb = 0; kb < D_DIM; kb += 32) {
        const int p = (kb >> 5) & 1;
        stage_wait();                              // tile p ready, buffers free
        if (kb + 32 < D_DIM) load_btile(kb + 32, p ^ 1);

        U16 af;
        const int oa = kb + 16 * hi;               // A-frag: K = oa..oa+15
        af.q[0] = *(const uint4*)(arow + oa);
        af.q[1] = *(const uint4*)(arow + oa + 8);

        const int ob = 16 * hi;                    // B-frag: K = ob..ob+15
        // Issue all 16 ds_load_b128 first (8 distinct register groups),
        // then the 8 WMMAs -> partial s_wait_dscnt, WMMA/DS overlap.
        U16 bfr[8];
#pragma unroll
        for (int t = 0; t < 8; ++t) {
            const __bf16* bp = &bstage[p][(t * 16 + lo) * 32 + ob];
            bfr[t].q[0] = *(const uint4*)bp;
            bfr[t].q[1] = *(const uint4*)(bp + 8);
        }
#pragma unroll
        for (int t = 0; t < 8; ++t)
            acc[t] = wmma_bf16(af.v, bfr[t].v, acc[t]);
    }

    // C layout: VGPR r, lane l -> row = r + 8*(l>>4), col = 16*t + (l&15)
#pragma unroll
    for (int t = 0; t < 8; ++t) {
        const int col = t * 16 + lo;
        const float b = bias[head * HD_DIM + col];
#pragma unroll
        for (int r = 0; r < 8; ++r)
            smem[(w * 16 + r + 8 * hi) * 128 + col] = acc[t][r] + b;
    }
    __syncthreads();

    if (z == 2) {
        // V: pack transposed -> vT[head][d][s]  (so PV B-frags are contiguous)
        const int col = tid;  // 0..127 == d
        __bf16* dst = vTo + ((size_t)head * HD_DIM + col) * S_LEN + blockIdx.x * 64;
        for (int row = 0; row < 64; ++row)
            dst[row] = (__bf16)smem[row * 128 + col];
        return;
    }

    // RMSNorm over HD=128 per row
    if (tid < 64) {
        float ss = 0.f;
        const float* rowp = &smem[tid * 128];
        for (int c = 0; c < 128; ++c) { float v = rowp[c]; ss += v * v; }
        sinv[tid] = rsqrtf(ss * (1.0f / 128.0f) + 1e-6f);
    }
    __syncthreads();

    const float* g = (z == 0) ? gq : gk;
    __bf16* dst = (z == 0) ? qo : ko;
    const int col = tid;             // 0..127
    const int pcol = col ^ 64;       // rotate_half partner
    const float gl = g[col], gp = g[pcol];
    for (int row = 0; row < 64; ++row) {
        const int s = blockIdx.x * 64 + row;
        const float inv = sinv[row];
        const float xn = smem[row * 128 + col] * inv * gl;
        const float xp = smem[row * 128 + pcol] * inv * gp;
        const float rot = (col < 64) ? -xp : xp;
        const float val = xn * cosb[s * 128 + col] + rot * sinb[s * 128 + col];
        dst[((size_t)head * S_LEN + s) * HD_DIM + col] = (__bf16)val;
    }
}

// ---------------------------------------------------------------------------
// Kernel 3: Flash attention. grid (32, 24): x = 64-query block, y = head.
// block 128 = 4 waves; wave owns 16 query rows and full 128-dim O accum.
// All 4 waves share each 32-key K-tile and V^T-tile -> stage both in LDS
// (double-buffered, async copies).
// ---------------------------------------------------------------------------
__global__ __launch_bounds__(128) void attn_kernel(
    const __bf16* __restrict__ qb, const __bf16* __restrict__ kbm,
    const __bf16* __restrict__ vT, float* __restrict__ out) {

    const int tid = threadIdx.x;
    const int l   = tid & 31;
    const int w   = tid >> 5;
    const int lo  = l & 15;
    const int hi  = l >> 4;
    const int head  = blockIdx.y;
    const int qbase = blockIdx.x * 64 + w * 16;

    __shared__ __bf16 kst[2][32 * 128];   // [key][d]   2 x 8KB
    __shared__ __bf16 vst[2][128 * 32];   // [d][s]     2 x 8KB
    __shared__ __bf16 smemP[4][16][32];   // per-wave P staging

    const __bf16* khead = kbm + (size_t)head * S_LEN * HD_DIM;
    const __bf16* vhead = vT + (size_t)head * HD_DIM * S_LEN;

    // Stage K tile (32 keys x 128) + V^T tile (128 x 32): 1024 chunks, 8/thread.
    auto load_kv = [&](int kb, int p) {
#pragma unroll
        for (int j = 0; j < 4; ++j) {       // K tile: 512 chunks
            const int chunk = tid + 128 * j;
            const int key   = chunk >> 4;   // 0..31
            const int part  = chunk & 15;   // 16 x 8 halves = 128 d
            cp16(khead + (size_t)(kb + key) * HD_DIM + part * 8,
                 &kst[p][key * 128 + part * 8]);
        }
#pragma unroll
        for (int j = 0; j < 4; ++j) {       // V tile: 512 chunks
            const int chunk = tid + 128 * j;
            const int d     = chunk >> 2;   // 0..127
            const int part  = chunk & 3;    // 4 x 8 halves = 32 s
            cp16(vhead + (size_t)d * S_LEN + kb + part * 8,
                 &vst[p][d * 32 + part * 8]);
        }
    };

    // Preload the wave's q A-fragments for the whole kernel (16 rows x 128).
    U16 qf[4];
    const __bf16* qrow = qb + ((size_t)head * S_LEN + (qbase + lo)) * HD_DIM;
#pragma unroll
    for (int f = 0; f < 4; ++f) {
        const int o = 32 * f + 16 * hi;
        qf[f].q[0] = *(const uint4*)(qrow + o);
        qf[f].q[1] = *(const uint4*)(qrow + o + 8);
    }

    v8f O[8];
#pragma unroll
    for (int t = 0; t < 8; ++t) O[t] = (v8f){};
    float m[8], lsum[8];
#pragma unroll
    for (int r = 0; r < 8; ++r) { m[r] = -1e30f; lsum[r] = 0.f; }

    const float scale = 0.08838834764831845f;  // 128^-0.5

    load_kv(0, 0);
    for (int kb = 0; kb < S_LEN; kb += 32) {
        const int p = (kb >> 5) & 1;
        stage_wait();                          // tiles p ready, buffers free
        if (kb + 32 < S_LEN) load_kv(kb + 32, p ^ 1);

        // ---- scores: S = q @ k^T  (two 16x16 N-tiles, K=128 via 4 frags)
        v8f s0 = (v8f){}, s1 = (v8f){};
#pragma unroll
        for (int f = 0; f < 4; ++f) {
            const int doff = 32 * f + 16 * hi;   // B-frag K slice
            const __bf16* kr0 = &kst[p][lo * 128 + doff];
            U16 b0; b0.q[0] = *(const uint4*)kr0; b0.q[1] = *(const uint4*)(kr0 + 8);
            s0 = wmma_bf16(qf[f].v, b0.v, s0);
            const __bf16* kr1 = &kst[p][(16 + lo) * 128 + doff];
            U16 b1; b1.q[0] = *(const uint4*)kr1; b1.q[1] = *(const uint4*)(kr1 + 8);
            s1 = wmma_bf16(qf[f].v, b1.v, s1);
        }

        // ---- online softmax per row (row r lives on a 16-lane group)
        float alpha[8];
#pragma unroll
        for (int r = 0; r < 8; ++r) {
            const float a = s0[r] * scale, b = s1[r] * scale;
            float mx = fmaxf(a, b);
#pragma unroll
            for (int off = 1; off < 16; off <<= 1)
                mx = fmaxf(mx, __shfl_xor(mx, off, 32));
            const float mn = fmaxf(m[r], mx);
            const float p0 = __expf(a - mn), p1 = __expf(b - mn);
            float rs = p0 + p1;
#pragma unroll
            for (int off = 1; off < 16; off <<= 1)
                rs += __shfl_xor(rs, off, 32);
            alpha[r] = __expf(m[r] - mn);
            lsum[r] = lsum[r] * alpha[r] + rs;
            m[r] = mn;
            smemP[w][r + 8 * hi][lo]      = (__bf16)p0;
            smemP[w][r + 8 * hi][16 + lo] = (__bf16)p1;
        }
#pragma unroll
        for (int t = 0; t < 8; ++t)
#pragma unroll
            for (int r = 0; r < 8; ++r) O[t][r] *= alpha[r];

        // ---- reload P as A-fragment (in-order DS within wave; private region)
        U16 pf;
        {
            const __bf16* prow = &smemP[w][lo][16 * hi];
            pf.q[0] = *(const uint4*)prow;
            pf.q[1] = *(const uint4*)(prow + 8);
        }

        // ---- O += P @ V   (V^T tile in LDS -> contiguous B-frags)
#pragma unroll
        for (int t = 0; t < 8; ++t) {
            const __bf16* vr = &vst[p][(t * 16 + lo) * 32 + 16 * hi];
            U16 vf; vf.q[0] = *(const uint4*)vr; vf.q[1] = *(const uint4*)(vr + 8);
            O[t] = wmma_bf16(pf.v, vf.v, O[t]);
        }
    }

    // ---- epilogue: O / l, scatter to [S, D] f32
#pragma unroll
    for (int r = 0; r < 8; ++r) {
        const float inv = 1.0f / lsum[r];
        const int s = qbase + r + 8 * hi;
#pragma unroll
        for (int t = 0; t < 8; ++t)
            out[(size_t)s * D_DIM + head * HD_DIM + t * 16 + lo] = O[t][r] * inv;
    }
}

// ---------------------------------------------------------------------------
// Launcher
// ---------------------------------------------------------------------------
extern "C" void kernel_launch(void* const* d_in, const int* in_sizes, int n_in,
                              void* d_out, int out_size, void* d_ws, size_t ws_size,
                              hipStream_t stream) {
    (void)in_sizes; (void)n_in; (void)out_size; (void)ws_size;

    const float* x    = (const float*)d_in[0];
    const float* Wq   = (const float*)d_in[1];
    const float* bq   = (const float*)d_in[2];
    const float* Wk   = (const float*)d_in[3];
    const float* bk   = (const float*)d_in[4];
    const float* Wv   = (const float*)d_in[5];
    const float* bv   = (const float*)d_in[6];
    const float* gq   = (const float*)d_in[7];
    const float* gk   = (const float*)d_in[8];
    const float* cosb = (const float*)d_in[9];
    const float* sinb = (const float*)d_in[10];
    float* out = (float*)d_out;

    const size_t XB = (size_t)S_LEN * D_DIM;     // 6,291,456
    const size_t WB = (size_t)D_DIM * D_DIM;     // 9,437,184
    const size_t HB = (size_t)H_NUM * S_LEN * HD_DIM;

    __bf16* ws  = (__bf16*)d_ws;
    __bf16* xb  = ws;
    __bf16* wqb = xb + XB;
    __bf16* wkb = wqb + WB;
    __bf16* wvb = wkb + WB;
    __bf16* qbf = wvb + WB;
    __bf16* kbf = qbf + HB;
    __bf16* vTb = kbf + HB;

    // 1) convert to bf16 (x + the three weight matrices)
    {
        int n4 = (int)(XB / 4);
        cvt4_kernel<<<(n4 + 255) / 256, 256, 0, stream>>>(x, xb, n4);
        n4 = (int)(WB / 4);
        cvt4_kernel<<<(n4 + 255) / 256, 256, 0, stream>>>(Wq, wqb, n4);
        cvt4_kernel<<<(n4 + 255) / 256, 256, 0, stream>>>(Wk, wkb, n4);
        cvt4_kernel<<<(n4 + 255) / 256, 256, 0, stream>>>(Wv, wvb, n4);
    }

    // 2) QKV GEMM + bias + RMSNorm + RoPE / V transpose-pack
    qkv_kernel<<<dim3(32, H_NUM, 3), 128, 0, stream>>>(
        xb, wqb, wkb, wvb, bq, bk, bv, gq, gk, cosb, sinb, qbf, kbf, vTb);

    // 3) flash attention
    attn_kernel<<<dim3(32, H_NUM), 128, 0, stream>>>(qbf, kbf, vTb, out);
}
